// MyViT_6184752906581
// MI455X (gfx1250) — matrix operand order repaired and checked by reading, not compiled, not run on hardware
//
#include <hip/hip_runtime.h>
#include <hip/hip_bf16.h>
#include <math.h>

typedef __attribute__((ext_vector_type(16))) _Float16 v16h;
typedef __attribute__((ext_vector_type(8)))  _Float16 v8h;
typedef __attribute__((ext_vector_type(8)))  float    v8f;
typedef __attribute__((ext_vector_type(2)))  float    v2f;

// Problem constants (match reference)
constexpr int HWD  = 28;
constexpr int NPP  = 7;
constexpr int PSZ  = 4;
constexpr int SS   = 50;    // tokens
constexpr int SP   = 64;    // padded tokens (4x 16-row tiles)
constexpr int DD   = 64;
constexpr int NHH  = 16;
constexpr int DHH  = 4;
constexpr int LL   = 6;
constexpr int DFF  = 256;
constexpr int IND  = 16;
constexpr int OUTC = 10;
constexpr float EPSF = 1e-5f;

// ---------------------------------------------------------------------------
// Prep: f32 -> f16 weight conversion with transpose into WMMA-B layout.
//   W1t[l][n][k] = W1[l][k][n]   (n: 0..255, k: 0..63)
//   W2t[l][n][k] = W2[l][k][n]   (n: 0..63,  k: 0..255)
// ---------------------------------------------------------------------------
__global__ __launch_bounds__(256)
void vit_prep_weights(const float* __restrict__ W1, const float* __restrict__ W2,
                      _Float16* __restrict__ W1t, _Float16* __restrict__ W2t) {
  int i = blockIdx.x * blockDim.x + threadIdx.x;
  const int per = LL * DD * DFF;
  if (i < per) {
    int l = i / (DD * DFF);
    int r = i % (DD * DFF);
    int n = r / DD;         // 0..255
    int k = r % DD;         // 0..63
    W1t[i] = (_Float16)W1[(l * DD + k) * DFF + n];
  } else if (i < 2 * per) {
    int j = i - per;
    int l = j / (DD * DFF);
    int r = j % (DD * DFF);
    int n = r / DFF;        // 0..63
    int k = r % DFF;        // 0..255
    W2t[j] = (_Float16)W2[(l * DFF + k) * DD + n];
  }
}

// ---------------------------------------------------------------------------
// Fused ViT forward: one workgroup (256 threads = 8 wave32) per image.
// Residual stream X[64][64] lives in LDS for all 6 layers.
// ---------------------------------------------------------------------------
__global__ __launch_bounds__(256)
void vit_forward(const float* __restrict__ images,
                 const float* __restrict__ Wp,   const float* __restrict__ bp,
                 const float* __restrict__ cls,  const float* __restrict__ pos,
                 const float* __restrict__ ln1g, const float* __restrict__ ln1b,
                 const float* __restrict__ Wq,   const float* __restrict__ bq,
                 const float* __restrict__ Wk,   const float* __restrict__ bk,
                 const float* __restrict__ Wv,   const float* __restrict__ bv,
                 const float* __restrict__ ln2g, const float* __restrict__ ln2b,
                 const float* __restrict__ b1,   const float* __restrict__ b2,
                 const float* __restrict__ Wh,   const float* __restrict__ bh,
                 const _Float16* __restrict__ W1t,
                 const _Float16* __restrict__ W2t,
                 float* __restrict__ out) {
  __shared__ float    X[SP * DD];                 // 16 KB residual (rows 50..63 junk-safe)
  __shared__ _Float16 Nf[SP * DD];                // 8 KB LN output (f16, WMMA A source)
  __shared__ __align__(16) char Ubuf[SP * DFF * 2]; // 32 KB union: Img | per-wave attn | Hf
  __shared__ float    Watt[3 * NHH * DHH * DHH];  // 3 KB  per-layer Wq/Wk/Wv
  __shared__ float    Batt[3 * NHH * DHH];        // 768 B per-layer bq/bk/bv
  __shared__ float    Lg[OUTC];

  float*    Img = (float*)Ubuf;                   // patchify phase
  _Float16* Hf  = (_Float16*)Ubuf;                // MLP phase [64][256] f16

  const int img  = blockIdx.x;
  const int tid  = threadIdx.x;
  const int lane = tid & 31;
  const int wave = tid >> 5;
  const int l15  = lane & 15;
  const int half8 = (lane & 16) ? 8 : 0;

  // ---- stage image, zero pad rows of X and Nf ----
  for (int i = tid; i < HWD * HWD; i += 256) Img[i] = images[img * HWD * HWD + i];
  for (int i = tid; i < (SP - SS) * DD; i += 256) {
    X[SS * DD + i]  = 0.0f;
    Nf[SS * DD + i] = (_Float16)0.0f;
  }
  __syncthreads();

  // ---- patch embed + cls + pos ----
  for (int i = tid; i < NPP * NPP * DD; i += 256) {
    int p = i / DD, d = i % DD;
    int pi = p / NPP, pj = p % NPP;
    float acc = bp[d];
#pragma unroll
    for (int e = 0; e < IND; ++e) {
      int ph = e / PSZ, pw = e % PSZ;
      acc += Img[(pi * PSZ + ph) * HWD + pj * PSZ + pw] * Wp[e * DD + d];
    }
    X[(1 + p) * DD + d] = acc + pos[(1 + p) * DD + d];
  }
  for (int i = tid; i < DD; i += 256) X[i] = cls[i] + pos[i];
  __syncthreads();

  const float scale = 0.5f;  // 1/sqrt(DH)

  for (int l = 0; l < LL; ++l) {
    const _Float16* W1l = W1t + l * DFF * DD;
    const _Float16* W2l = W2t + l * DD * DFF;
    if (lane == 0) {   // warm L2/L0 for this layer's f16 weights
      __builtin_prefetch(W1l, 0, 3);
      __builtin_prefetch(W2l, 0, 3);
    }

    // ---- stage attention weights for this layer ----
    for (int i = tid; i < NHH * DHH * DHH; i += 256) {
      Watt[i]                       = Wq[l * NHH * DHH * DHH + i];
      Watt[NHH * DHH * DHH + i]     = Wk[l * NHH * DHH * DHH + i];
      Watt[2 * NHH * DHH * DHH + i] = Wv[l * NHH * DHH * DHH + i];
    }
    for (int i = tid; i < NHH * DHH; i += 256) {
      Batt[i]                 = bq[l * NHH * DHH + i];
      Batt[NHH * DHH + i]     = bk[l * NHH * DHH + i];
      Batt[2 * NHH * DHH + i] = bv[l * NHH * DHH + i];
    }
    __syncthreads();

    // ---- LN1 -> Nf (f16) ----
    if (tid < SS) {
      const float* xr = &X[tid * DD];
      float mu = 0.f;
      for (int c = 0; c < DD; ++c) mu += xr[c];
      mu *= (1.0f / DD);
      float var = 0.f;
      for (int c = 0; c < DD; ++c) { float d = xr[c] - mu; var += d * d; }
      var *= (1.0f / DD);
      float rs = rsqrtf(var + EPSF);
      for (int c = 0; c < DD; ++c) {
        float y = (xr[c] - mu) * rs * ln1g[l * DD + c] + ln1b[l * DD + c];
        Nf[tid * DD + c] = (_Float16)y;
      }
    }
    __syncthreads();

    // ---- attention: per-wave, 2 heads/wave, all matmuls in WMMA ----
    {
      _Float16* Vt = (_Float16*)(Ubuf + wave * 4096);         // [16][64] f16: Vt[n][k]=v_n(tok k)
      _Float16* Aw = (_Float16*)(Ubuf + wave * 4096 + 2048);  // [16][64] f16 attn tile-row
      const int d0 = (lane & 16) ? 2 : 0;   // K-pair this lane holds (32-bit A/B frags)

      // zero Vt rows 4..15 (once per layer; Hf clobbered Ubuf last layer)
      {
        v8h z = {};
        for (int i = lane; i < 96; i += 32) ((v8h*)(Vt + 4 * 64))[i] = z;
      }
      __asm volatile("s_wait_dscnt 0x0" ::: "memory");

#pragma unroll
      for (int hh = 0; hh < 2; ++hh) {
        const int h = wave * 2 + hh;
        const float* wqh = &Watt[h * 16];
        const float* wkh = &Watt[NHH * DHH * DHH + h * 16];
        const float* wvh = &Watt[2 * NHH * DHH * DHH + h * 16];
        const float* bqh = &Batt[h * DHH];
        const float* bkh = &Batt[NHH * DHH + h * DHH];
        const float* bvh = &Batt[2 * NHH * DHH + h * DHH];

        // stage Vt rows 0..3: Vt[n][k] = v_n(token k)
        for (int i = lane; i < 4 * 64; i += 32) {
          int n = i >> 6, k = i & 63;
          float x0 = (float)Nf[k * DD + h * DHH + 0];
          float x1 = (float)Nf[k * DD + h * DHH + 1];
          float x2 = (float)Nf[k * DD + h * DHH + 2];
          float x3 = (float)Nf[k * DD + h * DHH + 3];
          float val = bvh[n] + x0 * wvh[0 * 4 + n] + x1 * wvh[1 * 4 + n]
                             + x2 * wvh[2 * 4 + n] + x3 * wvh[3 * 4 + n];
          Vt[n * 64 + k] = (_Float16)val;
        }
        __asm volatile("s_wait_dscnt 0x0" ::: "memory");

        // B fragments of K^T (4x16 f32) per nt tile, computed in registers
        v2f bf[4];
#pragma unroll
        for (int nt = 0; nt < 4; ++nt) {
          int n = nt * 16 + l15;
          float x0 = (float)Nf[n * DD + h * DHH + 0];
          float x1 = (float)Nf[n * DD + h * DHH + 1];
          float x2 = (float)Nf[n * DD + h * DHH + 2];
          float x3 = (float)Nf[n * DD + h * DHH + 3];
          bf[nt].x = bkh[d0]     + x0 * wkh[0 * 4 + d0]     + x1 * wkh[1 * 4 + d0]
                                 + x2 * wkh[2 * 4 + d0]     + x3 * wkh[3 * 4 + d0];
          bf[nt].y = bkh[d0 + 1] + x0 * wkh[0 * 4 + d0 + 1] + x1 * wkh[1 * 4 + d0 + 1]
                                 + x2 * wkh[2 * 4 + d0 + 1] + x3 * wkh[3 * 4 + d0 + 1];
        }

        const bool mask3 = (l15 >= 2);  // cols 48..63: only 48,49 are valid keys

#pragma unroll
        for (int mt = 0; mt < 4; ++mt) {
          int trow = mt * 16 + l15;
          float x0 = (float)Nf[trow * DD + h * DHH + 0];
          float x1 = (float)Nf[trow * DD + h * DHH + 1];
          float x2 = (float)Nf[trow * DD + h * DHH + 2];
          float x3 = (float)Nf[trow * DD + h * DHH + 3];
          v2f af;
          af.x = bqh[d0]     + x0 * wqh[0 * 4 + d0]     + x1 * wqh[1 * 4 + d0]
                             + x2 * wqh[2 * 4 + d0]     + x3 * wqh[3 * 4 + d0];
          af.y = bqh[d0 + 1] + x0 * wqh[0 * 4 + d0 + 1] + x1 * wqh[1 * 4 + d0 + 1]
                             + x2 * wqh[2 * 4 + d0 + 1] + x3 * wqh[3 * 4 + d0 + 1];

          v8f z = {};
          v8f c0 = __builtin_amdgcn_wmma_f32_16x16x4_f32(false, af, false, bf[0], (short)0, z, false, false);
          v8f c1 = __builtin_amdgcn_wmma_f32_16x16x4_f32(false, af, false, bf[1], (short)0, z, false, false);
          v8f c2 = __builtin_amdgcn_wmma_f32_16x16x4_f32(false, af, false, bf[2], (short)0, z, false, false);
          v8f c3 = __builtin_amdgcn_wmma_f32_16x16x4_f32(false, af, false, bf[3], (short)0, z, false, false);

          // masked, scaled softmax over 64 key columns (row = r+half8 within tile)
#pragma unroll
          for (int r = 0; r < 8; ++r) {
            float v0 = c0[r] * scale;
            float v1 = c1[r] * scale;
            float v2 = c2[r] * scale;
            float v3 = mask3 ? -1e30f : c3[r] * scale;
            float pm = fmaxf(fmaxf(v0, v1), fmaxf(v2, v3));
            pm = fmaxf(pm, __shfl_xor(pm, 1, 32));
            pm = fmaxf(pm, __shfl_xor(pm, 2, 32));
            pm = fmaxf(pm, __shfl_xor(pm, 4, 32));
            pm = fmaxf(pm, __shfl_xor(pm, 8, 32));
            float e0 = __expf(v0 - pm), e1 = __expf(v1 - pm);
            float e2 = __expf(v2 - pm), e3 = __expf(v3 - pm);
            float ps = e0 + e1 + e2 + e3;
            ps += __shfl_xor(ps, 1, 32);
            ps += __shfl_xor(ps, 2, 32);
            ps += __shfl_xor(ps, 4, 32);
            ps += __shfl_xor(ps, 8, 32);
            float inv = 1.0f / ps;
            int rr = r + half8;
            Aw[rr * 64 +  0 + l15] = (_Float16)(e0 * inv);
            Aw[rr * 64 + 16 + l15] = (_Float16)(e1 * inv);
            Aw[rr * 64 + 32 + l15] = (_Float16)(e2 * inv);
            Aw[rr * 64 + 48 + l15] = (_Float16)(e3 * inv);
          }
          __asm volatile("s_wait_dscnt 0x0" ::: "memory");

          // attn(16x64,f16) @ V(64x16 padded,f16) -> o(16x16), cols 0..3 valid
          v8f o = {};
#pragma unroll
          for (int ks = 0; ks < 2; ++ks) {
            int ka = ks * 32 + ((lane & 16) ? 8 : 0);
            v8h alo = *(const v8h*)&Aw[l15 * 64 + ka];
            v8h ahi = *(const v8h*)&Aw[l15 * 64 + ka + 16];
            v16h a = __builtin_shufflevector(alo, ahi, 0,1,2,3,4,5,6,7,8,9,10,11,12,13,14,15);
            int kb = ks * 32 + ((lane & 16) ? 8 : 0);
            v8h blo = *(const v8h*)&Vt[l15 * 64 + kb];
            v8h bhi = *(const v8h*)&Vt[l15 * 64 + kb + 16];
            v16h b = __builtin_shufflevector(blo, bhi, 0,1,2,3,4,5,6,7,8,9,10,11,12,13,14,15);
            o = __builtin_amdgcn_wmma_f32_16x16x32_f16(false, a, false, b, (short)0, o, false, false);
          }
          if (l15 < DHH) {
#pragma unroll
            for (int r = 0; r < 8; ++r)
              X[(mt * 16 + r + half8) * DD + h * DHH + l15] += o[r];
          }
        }
      }
    }
    __syncthreads();

    // ---- LN2 -> Nf (f16) ----
    if (tid < SS) {
      const float* xr = &X[tid * DD];
      float mu = 0.f;
      for (int c = 0; c < DD; ++c) mu += xr[c];
      mu *= (1.0f / DD);
      float var = 0.f;
      for (int c = 0; c < DD; ++c) { float d = xr[c] - mu; var += d * d; }
      var *= (1.0f / DD);
      float rs = rsqrtf(var + EPSF);
      for (int c = 0; c < DD; ++c) {
        float y = (xr[c] - mu) * rs * ln2g[l * DD + c] + ln2b[l * DD + c];
        Nf[tid * DD + c] = (_Float16)y;
      }
    }
    __syncthreads();

    // ---- GEMM1: Hf = gelu(Nf[64x64] @ W1[64x256] + b1)  (WMMA f16->f32) ----
    {
      const int kaofs = (lane & 16) ? 8 : 0;
      const int kbofs = (lane & 16) ? 8 : 0;
#pragma unroll
      for (int tt = 0; tt < 8; ++tt) {
        int tile = wave * 8 + tt;              // 64 tiles: 4 (M) x 16 (N)
        int mt = tile >> 4, nt = tile & 15;
        int row = mt * 16 + l15;
        int n   = nt * 16 + l15;
        v8f c = {};
#pragma unroll
        for (int ks = 0; ks < 2; ++ks) {
          int ka = ks * 32 + kaofs;
          v8h alo = *(const v8h*)&Nf[row * DD + ka];
          v8h ahi = *(const v8h*)&Nf[row * DD + ka + 16];
          v16h a = __builtin_shufflevector(alo, ahi, 0,1,2,3,4,5,6,7,8,9,10,11,12,13,14,15);
          int kb = ks * 32 + kbofs;
          const _Float16* bptr = W1l + n * DD + kb;
          v8h blo = *(const v8h*)bptr;
          v8h bhi = *(const v8h*)(bptr + 16);
          v16h b = __builtin_shufflevector(blo, bhi, 0,1,2,3,4,5,6,7,8,9,10,11,12,13,14,15);
          c = __builtin_amdgcn_wmma_f32_16x16x32_f16(false, a, false, b, (short)0, c, false, false);
        }
        float bias = b1[l * DFF + n];
#pragma unroll
        for (int r = 0; r < 8; ++r) {
          int mrow = mt * 16 + r + half8;
          float v = c[r] + bias;
          float g = 0.5f * v * (1.0f + erff(v * 0.70710678f));
          Hf[mrow * DFF + n] = (_Float16)g;
        }
      }
    }
    __syncthreads();

    // ---- GEMM2: X += Hf[64x256] @ W2[256x64] + b2  (WMMA f16->f32) ----
    {
      const int kaofs = (lane & 16) ? 8 : 0;
      const int kbofs = (lane & 16) ? 8 : 0;
#pragma unroll
      for (int tt = 0; tt < 2; ++tt) {
        int tile = wave * 2 + tt;              // 16 tiles: 4 (M) x 4 (N)
        int mt = tile >> 2, nt = tile & 3;
        int row = mt * 16 + l15;
        int n   = nt * 16 + l15;
        v8f c = {};
#pragma unroll
        for (int ks = 0; ks < 8; ++ks) {
          int ka = ks * 32 + kaofs;
          v8h alo = *(const v8h*)&Hf[row * DFF + ka];
          v8h ahi = *(const v8h*)&Hf[row * DFF + ka + 16];
          v16h a = __builtin_shufflevector(alo, ahi, 0,1,2,3,4,5,6,7,8,9,10,11,12,13,14,15);
          int kb = ks * 32 + kbofs;
          const _Float16* bptr = W2l + n * DFF + kb;
          v8h blo = *(const v8h*)bptr;
          v8h bhi = *(const v8h*)(bptr + 16);
          v16h b = __builtin_shufflevector(blo, bhi, 0,1,2,3,4,5,6,7,8,9,10,11,12,13,14,15);
          c = __builtin_amdgcn_wmma_f32_16x16x32_f16(false, a, false, b, (short)0, c, false, false);
        }
        float bias = b2[l * DD + n];
#pragma unroll
        for (int r = 0; r < 8; ++r) {
          int mrow = mt * 16 + r + half8;
          X[mrow * DD + n] += c[r] + bias;
        }
      }
    }
    __syncthreads();
  }

  // ---- head: logits = X[0] @ Wh + bh ; softmax ----
  if (tid < OUTC) {
    float acc = bh[tid];
    for (int k = 0; k < DD; ++k) acc += X[k] * Wh[k * OUTC + tid];
    Lg[tid] = acc;
  }
  __syncthreads();
  if (tid == 0) {
    float mx = Lg[0];
    for (int j = 1; j < OUTC; ++j) mx = fmaxf(mx, Lg[j]);
    float sum = 0.f;
    float e[OUTC];
    for (int j = 0; j < OUTC; ++j) { e[j] = __expf(Lg[j] - mx); sum += e[j]; }
    float inv = 1.0f / sum;
    for (int j = 0; j < OUTC; ++j) out[img * OUTC + j] = e[j] * inv;
  }
}

// ---------------------------------------------------------------------------
extern "C" void kernel_launch(void* const* d_in, const int* in_sizes, int n_in,
                              void* d_out, int out_size, void* d_ws, size_t ws_size,
                              hipStream_t stream) {
  const float* images = (const float*)d_in[0];
  const float* Wp     = (const float*)d_in[1];
  const float* bp     = (const float*)d_in[2];
  const float* cls    = (const float*)d_in[3];
  const float* pos    = (const float*)d_in[4];
  const float* ln1g   = (const float*)d_in[5];
  const float* ln1b   = (const float*)d_in[6];
  const float* Wq     = (const float*)d_in[7];
  const float* bq     = (const float*)d_in[8];
  const float* Wk     = (const float*)d_in[9];
  const float* bk     = (const float*)d_in[10];
  const float* Wv     = (const float*)d_in[11];
  const float* bv     = (const float*)d_in[12];
  const float* ln2g   = (const float*)d_in[13];
  const float* ln2b   = (const float*)d_in[14];
  const float* W1     = (const float*)d_in[15];
  const float* b1     = (const float*)d_in[16];
  const float* W2     = (const float*)d_in[17];
  const float* b2     = (const float*)d_in[18];
  const float* Wh     = (const float*)d_in[19];
  const float* bh     = (const float*)d_in[20];
  float* out = (float*)d_out;

  _Float16* W1t = (_Float16*)d_ws;
  _Float16* W2t = W1t + LL * DFF * DD;

  const int nimg = in_sizes[0] / (HWD * HWD);
  const int prepN = 2 * LL * DD * DFF;

  hipLaunchKernelGGL(vit_prep_weights, dim3((prepN + 255) / 256), dim3(256), 0, stream,
                     W1, W2, W1t, W2t);
  hipLaunchKernelGGL(vit_forward, dim3(nimg), dim3(256), 0, stream,
                     images, Wp, bp, cls, pos, ln1g, ln1b,
                     Wq, bq, Wk, bk, Wv, bv, ln2g, ln2b,
                     b1, b2, Wh, bh, W1t, W2t, out);
}
